// SpatialCrossAttention_49228915147512
// MI455X (gfx1250) — compile-verified, matrix operand order checked by reading, not run on hardware
//
#include <hip/hip_runtime.h>
#include <hip/hip_fp16.h>
#include <math.h>

typedef float    v2f    __attribute__((ext_vector_type(2)));
typedef float    v4f    __attribute__((ext_vector_type(4)));
typedef float    v8f    __attribute__((ext_vector_type(8)));
typedef _Float16 f16x16 __attribute__((ext_vector_type(16)));

#define HF   56
#define WF   100
#define CCH  256
#define NCAM 6

// ---------------------------------------------------------------------------
// Kernel 0: transpose value (N, C, H*W) -> (N, H*W, C) so that the bilinear
// gather reads 256 contiguous floats per corner (1KB coalesced lines, L2-hot).
// H*W = 5600 = 175*32, C = 256 = 8*32 -> exact 32x32 tiles, no bounds checks.
// ---------------------------------------------------------------------------
__global__ void __launch_bounds__(256)
sca_transpose_value(const float* __restrict__ v, float* __restrict__ vT)
{
    __shared__ float tile[32][33];
    const int HW = HF * WF;
    const int n  = blockIdx.z;
    const int c0 = blockIdx.y * 32;
    const int p0 = blockIdx.x * 32;
    #pragma unroll
    for (int i = 0; i < 32; i += 8) {
        int c = c0 + threadIdx.y + i;
        int p = p0 + threadIdx.x;
        tile[threadIdx.y + i][threadIdx.x] = v[((size_t)n * CCH + c) * HW + p];
    }
    __syncthreads();
    #pragma unroll
    for (int i = 0; i < 32; i += 8) {
        int p = p0 + threadIdx.y + i;
        int c = c0 + threadIdx.x;
        vT[((size_t)n * HW + p) * CCH + c] = tile[threadIdx.x][threadIdx.y + i];
    }
}

// ---------------------------------------------------------------------------
// Kernel 1: fused grid-sample + LayerNorm + masked online-softmax attention.
// One wave32 per query l; lane owns the CONTIGUOUS channels [lane*8, lane*8+8)
// so every corner fetch is two global_load_b128 (wave covers the same 1KB).
// Never materializes the (6, L, 256) sampled tensor (online softmax).
// ---------------------------------------------------------------------------
__global__ void __launch_bounds__(256)
sca_sample_attend(const float* __restrict__ query,   // (L, 256)
                  const float* __restrict__ vT,      // (N, H*W, 256)
                  const float* __restrict__ grid,    // (N, L, 2)
                  const int*   __restrict__ mask,    // (N, L)
                  float*       __restrict__ tmp,     // (L, 256)
                  int L)
{
    const int lane = threadIdx.x & 31;
    const int wave = threadIdx.x >> 5;
    const int l = blockIdx.x * 8 + wave;
    if (l >= L) return;

    const int chbase = lane * 8;
    const v4f* qp = (const v4f*)(query + (size_t)l * CCH + chbase);
    v4f qa = qp[0], qb = qp[1];

    v4f acc0 = {}, acc1 = {};     // online-softmax weighted accumulator
    v4f sum0 = {}, sum1 = {};     // plain sum (for the all-invalid uniform path)
    float mrun = -__builtin_inff();
    float Z = 0.f;
    int nvalid = 0;

    for (int n = 0; n < NCAM; ++n) {
        // unnormalize (align_corners=False) + clamp to border; wave-uniform.
        float gx = grid[((size_t)n * L + l) * 2 + 0];
        float gy = grid[((size_t)n * L + l) * 2 + 1];
        float ix = fminf(fmaxf(((gx + 1.f) * (float)WF - 1.f) * 0.5f, 0.f), (float)(WF - 1));
        float iy = fminf(fmaxf(((gy + 1.f) * (float)HF - 1.f) * 0.5f, 0.f), (float)(HF - 1));
        float x0f = floorf(ix), y0f = floorf(iy);
        float wx = ix - x0f,   wy = iy - y0f;
        int x0 = (int)x0f, y0 = (int)y0f;
        int x1 = min(x0 + 1, WF - 1);
        int y1 = min(y0 + 1, HF - 1);
        const float* base = vT + (size_t)n * (HF * WF) * CCH + chbase;
        const float* p00 = base + ((size_t)y0 * WF + x0) * CCH;
        const float* p01 = base + ((size_t)y0 * WF + x1) * CCH;
        const float* p10 = base + ((size_t)y1 * WF + x0) * CCH;
        const float* p11 = base + ((size_t)y1 * WF + x1) * CCH;
        float w00 = (1.f - wx) * (1.f - wy);
        float w01 = wx * (1.f - wy);
        float w10 = (1.f - wx) * wy;
        float w11 = wx * wy;

        v4f A00 = ((const v4f*)p00)[0], B00 = ((const v4f*)p00)[1];
        v4f A01 = ((const v4f*)p01)[0], B01 = ((const v4f*)p01)[1];
        v4f A10 = ((const v4f*)p10)[0], B10 = ((const v4f*)p10)[1];
        v4f A11 = ((const v4f*)p11)[0], B11 = ((const v4f*)p11)[1];

        v4f sa = w00 * A00 + w01 * A01 + w10 * A10 + w11 * A11;
        v4f sb = w00 * B00 + w01 * B01 + w10 * B10 + w11 * B11;

        float lsum = sa.x + sa.y + sa.z + sa.w + sb.x + sb.y + sb.z + sb.w;
        float lsq  = sa.x*sa.x + sa.y*sa.y + sa.z*sa.z + sa.w*sa.w
                   + sb.x*sb.x + sb.y*sb.y + sb.z*sb.z + sb.w*sb.w;
        // wave32 reductions for LayerNorm statistics
        #pragma unroll
        for (int off = 16; off > 0; off >>= 1) {
            lsum += __shfl_xor(lsum, off, 32);
            lsq  += __shfl_xor(lsq,  off, 32);
        }
        float mu   = lsum * (1.f / 256.f);
        float var  = lsq * (1.f / 256.f) - mu * mu;
        float rstd = rsqrtf(var + 1e-5f);
        sa = (sa - mu) * rstd;
        sb = (sb - mu) * rstd;
        sum0 += sa;
        sum1 += sb;

        float dot = qa.x*sa.x + qa.y*sa.y + qa.z*sa.z + qa.w*sa.w
                  + qb.x*sb.x + qb.y*sb.y + qb.z*sb.z + qb.w*sb.w;
        #pragma unroll
        for (int off = 16; off > 0; off >>= 1) dot += __shfl_xor(dot, off, 32);
        float score = dot * 0.0625f;      // 1/sqrt(256)

        if (mask[(size_t)n * L + l] != 0) {   // wave-uniform branch
            ++nvalid;
            float mnew = fmaxf(mrun, score);
            float corr = __expf(mrun - mnew); // exp(-inf)=0 on first valid cam
            float p    = __expf(score - mnew);
            Z = Z * corr + p;
            acc0 = acc0 * corr + p * sa;
            acc1 = acc1 * corr + p * sb;
            mrun = mnew;
        }
    }

    if (nvalid == 0) {
        acc0 = sum0 * (1.f / 6.f);
        acc1 = sum1 * (1.f / 6.f);
    } else {
        float inv = 1.f / Z;
        acc0 *= inv;
        acc1 *= inv;
    }
    v4f* op = (v4f*)(tmp + (size_t)l * CCH + chbase);
    op[0] = acc0;
    op[1] = acc1;
}

// ---------------------------------------------------------------------------
// Kernel 2: output projection  out = tmp(L,256) @ W_out^T(256,256) + b_out.
// One wave32 per 16x16 output tile, WMMA f32 16x16x4 over K=256.
// A fragment (32-bit A 16x4 layout): lane m = lane&15, K pair = k + 2*(lane>>4).
// B fragment symmetric with B[k][n] = W_out[n][k] -> contiguous reads of row n.
// C/D: 8 VGPRs, n = lane&15, m = r + 8*(lane>>4). Bias folded into C init.
// ---------------------------------------------------------------------------
__global__ void __launch_bounds__(256)
sca_out_proj(const float* __restrict__ A,     // (L, 256)
             const float* __restrict__ Wout,  // (256, 256) row j = weights for out col j
             const float* __restrict__ bias,  // (256)
             float*       __restrict__ out,   // (L, 256)
             int L)
{
    const int lane = threadIdx.x & 31;
    const int wave = threadIdx.x >> 5;
    const int mtiles = (L + 15) >> 4;
    int tile = blockIdx.x * 8 + wave;
    if (tile >= mtiles * 16) return;
    const int ntile = tile & 15;
    const int mtile = tile >> 4;
    const int m0 = mtile * 16;
    const int n0 = ntile * 16;
    const int lid  = lane & 15;
    const int half = lane >> 4;

    int am = m0 + lid; if (am >= L) am = L - 1;
    const float* arow = A    + (size_t)am * CCH;
    const float* brow = Wout + (size_t)(n0 + lid) * CCH;

    v8f acc;
    {
        float bv = bias[n0 + lid];          // bias depends only on n -> fold into C
        #pragma unroll
        for (int r = 0; r < 8; ++r) acc[r] = bv;
    }

#if __has_builtin(__builtin_amdgcn_wmma_f32_16x16x4_f32)
    #pragma unroll 8
    for (int k = 0; k < CCH; k += 4) {
        int ka = k + half * 2;
        v2f a = *(const v2f*)(arow + ka);
        v2f b = *(const v2f*)(brow + ka);
        acc = __builtin_amdgcn_wmma_f32_16x16x4_f32(
            false, a, false, b, (short)0, acc, false, false);
    }
#elif __has_builtin(__builtin_amdgcn_wmma_f32_16x16x32_f16)
    for (int k = 0; k < CCH; k += 32) {
        f16x16 a, b;
        #pragma unroll
        for (int e = 0; e < 16; ++e) {
            int vg = e >> 1, slot = e & 1;
            int kk = (vg < 4) ? (8 * half + 2 * vg + slot)
                              : (16 + 8 * half + 2 * (vg - 4) + slot);
            a[e] = (_Float16)arow[k + kk];
            b[e] = (_Float16)brow[k + kk];
        }
        acc = __builtin_amdgcn_wmma_f32_16x16x32_f16(
            false, a, false, b, (short)0, acc, false, false);
    }
#else
    #pragma unroll
    for (int r = 0; r < 8; ++r) {
        int m = m0 + r + half * 8; if (m >= L) m = L - 1;
        const float* ar = A + (size_t)m * CCH;
        float d = 0.f;
        for (int k = 0; k < CCH; ++k) d += ar[k] * brow[k];
        acc[r] += d;
    }
#endif

    const int mbase = m0 + half * 8;
    float* obase = out + (size_t)mbase * CCH + n0 + lid;
    if (mbase + 8 <= L) {
        // fast path: full tile in range (always taken when L % 16 == 0)
        #pragma unroll
        for (int r = 0; r < 8; ++r) obase[(size_t)r * CCH] = acc[r];
    } else {
        #pragma unroll
        for (int r = 0; r < 8; ++r) {
            if (mbase + r < L) obase[(size_t)r * CCH] = acc[r];
        }
    }
}

// ---------------------------------------------------------------------------
extern "C" void kernel_launch(void* const* d_in, const int* in_sizes, int n_in,
                              void* d_out, int out_size, void* d_ws, size_t ws_size,
                              hipStream_t stream)
{
    (void)n_in; (void)out_size; (void)ws_size;
    const float* query = (const float*)d_in[0];   // (1, L, 256)
    const float* value = (const float*)d_in[2];   // (6, 256, 56, 100)
    const float* refp  = (const float*)d_in[4];   // (1, 6, L, 2)
    const int*   bmask = (const int*)d_in[5];     // (1, 6, L)
    const float* Wout  = (const float*)d_in[8];   // (256, 256)
    const float* bout  = (const float*)d_in[9];   // (256)
    float* out = (float*)d_out;

    const int L = in_sizes[0] / CCH;              // 40000

    // workspace: vT (34.4 MB) + tmp (L*256*4 bytes); both fully overwritten.
    float* vT  = (float*)d_ws;
    float* tmp = vT + (size_t)NCAM * HF * WF * CCH;

    dim3 tblock(32, 8, 1);
    dim3 tgrid((HF * WF) / 32, CCH / 32, NCAM);
    sca_transpose_value<<<tgrid, tblock, 0, stream>>>(value, vT);

    int qblocks = (L + 7) / 8;
    sca_sample_attend<<<qblocks, 256, 0, stream>>>(query, vT, refp, bmask, tmp, L);

    int mtiles = (L + 15) / 16;
    int gblocks = (mtiles * 16 + 7) / 8;
    sca_out_proj<<<gblocks, 256, 0, stream>>>(tmp, Wout, bout, out, L);
}
